// MiMoV2FlashAttention_17437567222555
// MI455X (gfx1250) — compile-verified, matrix operand order checked by reading, not run on hardware
//
#include <hip/hip_runtime.h>

typedef __bf16 bf16;
typedef __attribute__((ext_vector_type(16))) __bf16 v16bf;
typedef __attribute__((ext_vector_type(8)))  float  v8f;
typedef __attribute__((ext_vector_type(4)))  unsigned int   u32x4;
typedef __attribute__((ext_vector_type(4)))  unsigned short u16x4;
typedef __attribute__((ext_vector_type(4)))  float          f32x4;

#define Bx   2
#define Sx   1024
#define HIDx 4096
#define NHx  64
#define NKVx 8
#define HDx  192
#define VDx  128
#define WINx 128
#define SCALEx 0.07216878364870323f   // 192^-0.5

static __device__ __forceinline__ unsigned short f2bfu(float f) {
  unsigned u = __builtin_bit_cast(unsigned, f);
  return (unsigned short)((u + 0x7FFFu + ((u >> 16) & 1u)) >> 16);
}
static __device__ __forceinline__ bf16 f2bf(float f) {
  unsigned short s = f2bfu(f);
  return __builtin_bit_cast(bf16, s);
}
static __device__ __forceinline__ float bf2f(bf16 h) {
  unsigned u = ((unsigned)__builtin_bit_cast(unsigned short, h)) << 16;
  return __builtin_bit_cast(float, u);
}

union FragBF { v16bf v; u32x4 u[2]; };
union U8     { u32x4 u; bf16 h[8]; };

static __device__ __forceinline__ v8f wmma_bf(v16bf a, v16bf b, v8f c) {
  // D = A(16x32 bf16) * B(32x16 bf16) + C(16x16 f32)
  return __builtin_amdgcn_wmma_f32_16x16x32_bf16(false, a, false, b, (short)0, c, false, false);
}

// ---------------------------------------------------------------- fp32->bf16
__global__ void cvt_kernel(const float* __restrict__ in, bf16* __restrict__ out, int n4) {
  int i = blockIdx.x * blockDim.x + threadIdx.x;
  if (i >= n4) return;
  f32x4 v = ((const f32x4*)in)[i];
  u16x4 o;
  o.x = f2bfu(v.x); o.y = f2bfu(v.y); o.z = f2bfu(v.z); o.w = f2bfu(v.w);
  ((u16x4*)out)[i] = o;
}

// ------------------------------------------------------- bf16 GEMM, 128x128 tile
// C[M,N] = A[M,K] * B[K,N]; M,N multiples of 128, K multiple of 64.
// Double-buffered LDS, K-step 64. A tile staged with async global->LDS copies
// (ASYNCcnt), B tile transposed via packed-dword DS stores.
template <typename CT>
__global__ __launch_bounds__(256) void gemm_bf16_tile(
    const bf16* __restrict__ A, const bf16* __restrict__ Bm,
    CT* __restrict__ C, int M, int N, int K)
{
  __shared__ __attribute__((aligned(16))) bf16 As[2][128][72];  // [buf][m][k64+pad]
  __shared__ __attribute__((aligned(16))) bf16 BT[2][128][72];  // [buf][n][k64+pad]
  const int t = threadIdx.x;
  const int wave = t >> 5, lane = t & 31;
  const int lm = lane & 15, lh = lane >> 4;
  const int wm = wave & 3, wn = wave >> 2;          // 4 waves in M, 2 in N
  const int m0 = blockIdx.y * 128, n0 = blockIdx.x * 128;

  v8f acc[2][4];
#pragma unroll
  for (int i = 0; i < 2; ++i)
#pragma unroll
    for (int j = 0; j < 4; ++j) acc[i][j] = {};

  const int arow = t >> 1, acol = (t & 1) << 5;     // A: 128 rows x 64 k, 64B/thread
  const bf16* aptr = A + (size_t)(m0 + arow) * K + acol;

  // ---- stage one 64-wide K tile into LDS buffer `buf`
  auto stage = [&](int k0, int buf) {
    // A tile: verbatim copy -> async global->LDS, 4 x 16B per thread
    unsigned lbase = (unsigned)(size_t)&As[buf][arow][acol];
    const bf16* ga = aptr + k0;
#pragma unroll
    for (int c = 0; c < 4; ++c) {
      unsigned ldst = lbase + (unsigned)(c * 16);
      unsigned long long gsrc = (unsigned long long)(size_t)(ga + c * 8);
      asm volatile("global_load_async_to_lds_b128 %0, %1, off"
                   :: "v"(ldst), "v"(gsrc) : "memory");
    }
    // B tile: 64 rows x 128 cols, transpose into BT[n][k]; write k-pairs as dwords
#pragma unroll
    for (int i = 0; i < 2; ++i) {
      const int id = t + (i << 8);
      const int kp = id >> 4;          // 0..31 : pair of k rows (2kp, 2kp+1)
      const int nc = id & 15;          // 0..15 : 8-column chunk
      const bf16* gb = Bm + (size_t)(k0 + 2 * kp) * N + n0 + nc * 8;
      U8 r0, r1;
      r0.u = *(const u32x4*)gb;
      r1.u = *(const u32x4*)(gb + N);
#pragma unroll
      for (int e = 0; e < 8; ++e) {
        unsigned pair = (unsigned)__builtin_bit_cast(unsigned short, r0.h[e])
                      | ((unsigned)__builtin_bit_cast(unsigned short, r1.h[e]) << 16);
        *(unsigned*)&BT[buf][nc * 8 + e][2 * kp] = pair;
      }
    }
  };

  // ---- consume one 64-wide K tile from LDS buffer `buf`
  auto compute = [&](int buf) {
#pragma unroll
    for (int ks = 0; ks < 64; ks += 32) {
      FragBF af[2], bfh[4];
#pragma unroll
      for (int mf = 0; mf < 2; ++mf) {
        const int r = wm * 32 + mf * 16 + lm;
        af[mf].u[0] = *(const u32x4*)&As[buf][r][ks + 8 * lh];       // K = 8h..
        af[mf].u[1] = *(const u32x4*)&As[buf][r][ks + 16 + 8 * lh];  // K = 16+8h..
      }
#pragma unroll
      for (int nf = 0; nf < 4; ++nf) {
        const int c = wn * 64 + nf * 16 + lm;
        bfh[nf].u[0] = *(const u32x4*)&BT[buf][c][ks + 8 * lh];
        bfh[nf].u[1] = *(const u32x4*)&BT[buf][c][ks + 16 + 8 * lh];
      }
#pragma unroll
      for (int mf = 0; mf < 2; ++mf)
#pragma unroll
        for (int nf = 0; nf < 4; ++nf)
          acc[mf][nf] = wmma_bf(af[mf].v, bfh[nf].v, acc[mf][nf]);
    }
  };

  const int nk = K >> 6;
  stage(0, 0);
  asm volatile("s_wait_asynccnt 0x0" ::: "memory");
  __syncthreads();
  for (int kb = 0; kb < nk; ++kb) {
    if (kb + 1 < nk) stage((kb + 1) << 6, (kb + 1) & 1);  // prefetch next buffer
    compute(kb & 1);
    asm volatile("s_wait_asynccnt 0x0" ::: "memory");
    __syncthreads();
  }

#pragma unroll
  for (int mf = 0; mf < 2; ++mf)
#pragma unroll
    for (int nf = 0; nf < 4; ++nf)
#pragma unroll
      for (int r = 0; r < 8; ++r) {
        int row = m0 + wm * 32 + mf * 16 + r + 8 * lh;
        int col = n0 + wn * 64 + nf * 16 + lm;
        float vv = acc[mf][nf][r];
        if constexpr (sizeof(CT) == sizeof(float)) C[(size_t)row * N + col] = vv;
        else                                       C[(size_t)row * N + col] = f2bf(vv);
      }
}

// ---------------------------------------------------------------- partial RoPE
__global__ void rope_kernel(bf16* __restrict__ X, const float* __restrict__ cs,
                            const float* __restrict__ sn, int nheads)
{
  int t = blockIdx.x * blockDim.x + threadIdx.x;
  int i  = t & 31;
  int hd = (t >> 5) % nheads;
  int bs = (t >> 5) / nheads;
  if (bs >= Bx * Sx) return;
  bf16* p = X + ((size_t)bs * nheads + hd) * HDx;
  float x1 = bf2f(p[i]), x2 = bf2f(p[i + 32]);
  float c1 = cs[(size_t)bs * 64 + i],      s1 = sn[(size_t)bs * 64 + i];
  float c2 = cs[(size_t)bs * 64 + i + 32], s2 = sn[(size_t)bs * 64 + i + 32];
  p[i]      = f2bf(x1 * c1 - x2 * s1);
  p[i + 32] = f2bf(x2 * c2 + x1 * s2);
}

// ------------------------------------------- sliding-window attention w/ sinks
__global__ __launch_bounds__(256) void attn_kernel(
    const bf16* __restrict__ Q, const bf16* __restrict__ Kb,
    const bf16* __restrict__ V, const float* __restrict__ sinks,
    bf16* __restrict__ O)
{
  __shared__ __attribute__((aligned(16))) bf16 VT[128][40];  // [vd][k], upper k half zero
  __shared__ __attribute__((aligned(16))) bf16 P[8][16][40]; // per-wave probs [q][k]
  const int t = threadIdx.x, wave = t >> 5, lane = t & 31;
  const int lm = lane & 15, lh = lane >> 4;
  const int b = blockIdx.z, h = blockIdx.y, q0 = blockIdx.x * 128;
  const int hkv = h >> 3;   // NH/NKV = 8

  {
    bf16 z = f2bf(0.f);
    bf16* vz = &VT[0][0];
    for (int i = t; i < 128 * 40; i += 256) vz[i] = z;
    bf16* pz = &P[0][0][0];
    for (int i = t; i < 8 * 16 * 40; i += 256) pz[i] = z;
  }
  __syncthreads();

  // Q fragments for this wave's 16 query rows: 6 K-chunks of 32 over HD=192
  const int qrow = q0 + wave * 16 + lm;
  const bf16* qp = Q + ((size_t)b * Sx + qrow) * (NHx * HDx) + h * HDx;
  FragBF qf[6];
#pragma unroll
  for (int d = 0; d < 6; ++d) {
    qf[d].u[0] = *(const u32x4*)(qp + d * 32 + 8 * lh);
    qf[d].u[1] = *(const u32x4*)(qp + d * 32 + 16 + 8 * lh);
  }

  const float sv = sinks[h];         // sink participates as an extra logit
  float mrun[8], lrun[8];
  v8f oacc[8];
#pragma unroll
  for (int r = 0; r < 8; ++r) { mrun[r] = sv; lrun[r] = 1.0f; }
#pragma unroll
  for (int tt = 0; tt < 8; ++tt) oacc[tt] = {};

  for (int kt = 0; kt < 16; ++kt) {
    const int ks = q0 - 128 + kt * 16;   // uniform across block
    __syncthreads();
    if (ks >= 0) {                        // stage V tile transposed: VT[vd][k]
      const int krow = t >> 4, ch = t & 15;
      U8 vv;
      vv.u = *(const u32x4*)(V + ((size_t)b * Sx + ks + krow) * (NKVx * VDx)
                               + hkv * VDx + ch * 8);
#pragma unroll
      for (int e = 0; e < 8; ++e) VT[ch * 8 + e][krow] = vv.h[e];
    }
    __syncthreads();
    if (ks < 0) continue;

    // S = Q * K^T : B-frag lane = key column, contiguous per-row K loads
    const bf16* kp = Kb + ((size_t)b * Sx + ks + lm) * (NKVx * HDx) + hkv * HDx;
    v8f s = {};
#pragma unroll
    for (int d = 0; d < 6; ++d) {
      FragBF kf;
      kf.u[0] = *(const u32x4*)(kp + d * 32 + 8 * lh);
      kf.u[1] = *(const u32x4*)(kp + d * 32 + 16 + 8 * lh);
      s = wmma_bf(qf[d].v, kf.v, s);
    }

    // mask + online softmax (row = query, spread over 16 lanes per half)
#pragma unroll
    for (int r = 0; r < 8; ++r) {
      const int row = q0 + wave * 16 + r + 8 * lh;
      const int col = ks + lm;
      float sc = s[r] * SCALEx;
      bool ok = (col <= row) && (row - col < WINx);
      sc = ok ? sc : -3.0e38f;
      float m = sc;
#pragma unroll
      for (int dd = 1; dd < 16; dd <<= 1) m = fmaxf(m, __shfl_xor(m, dd, 32));
      float mn = fmaxf(mrun[r], m);
      float al = __expf(mrun[r] - mn);
      float pe = __expf(sc - mn);
      float rs = pe;
#pragma unroll
      for (int dd = 1; dd < 16; dd <<= 1) rs += __shfl_xor(rs, dd, 32);
      lrun[r] = lrun[r] * al + rs;
      mrun[r] = mn;
#pragma unroll
      for (int tt = 0; tt < 8; ++tt) oacc[tt][r] *= al;
      P[wave][r + 8 * lh][lm] = f2bf(pe);   // transpose via LDS
    }

    // O += P(16x32, upper 16 keys zero) * V(32x128)
    FragBF pf;
    pf.u[0] = *(const u32x4*)&P[wave][lm][8 * lh];
    pf.u[1] = *(const u32x4*)&P[wave][lm][16 + 8 * lh];
#pragma unroll
    for (int tt = 0; tt < 8; ++tt) {
      FragBF vf;
      vf.u[0] = *(const u32x4*)&VT[tt * 16 + lm][8 * lh];
      vf.u[1] = *(const u32x4*)&VT[tt * 16 + lm][16 + 8 * lh];
      oacc[tt] = wmma_bf(pf.v, vf.v, oacc[tt]);
    }
  }

  bf16* op = O + ((size_t)b * Sx + q0 + wave * 16) * (NHx * VDx) + h * VDx;
#pragma unroll
  for (int tt = 0; tt < 8; ++tt)
#pragma unroll
    for (int r = 0; r < 8; ++r) {
      float val = oacc[tt][r] / lrun[r];
      op[(size_t)(r + 8 * lh) * (NHx * VDx) + tt * 16 + lm] = f2bf(val);
    }
}

// ---------------------------------------------------------------------- launch
extern "C" void kernel_launch(void* const* d_in, const int* in_sizes, int n_in,
                              void* d_out, int out_size, void* d_ws, size_t ws_size,
                              hipStream_t stream)
{
  (void)in_sizes; (void)n_in; (void)out_size; (void)ws_size;
  const float* hs    = (const float*)d_in[0];
  const float* cosb  = (const float*)d_in[1];
  const float* sinb  = (const float*)d_in[2];
  const float* Wq    = (const float*)d_in[3];
  const float* Wk    = (const float*)d_in[4];
  const float* Wv    = (const float*)d_in[5];
  const float* Wo    = (const float*)d_in[6];
  const float* sinks = (const float*)d_in[7];
  float* out = (float*)d_out;

  const size_t BS = (size_t)Bx * Sx;   // 2048
  char* w = (char*)d_ws;
  auto take = [&](size_t elems) {
    bf16* p = (bf16*)w;
    w += ((elems * 2) + 255) & ~(size_t)255;
    return p;
  };
  bf16* hid_bf = take(BS * HIDx);
  bf16* wq_bf  = take((size_t)HIDx * NHx * HDx);
  bf16* wk_bf  = take((size_t)HIDx * NKVx * HDx);
  bf16* wv_bf  = take((size_t)HIDx * NKVx * VDx);
  bf16* wo_bf  = take((size_t)NHx * VDx * HIDx);
  bf16* Qb     = take(BS * NHx * HDx);
  bf16* Kbuf   = take(BS * NKVx * HDx);
  bf16* Vb     = take(BS * NKVx * VDx);
  bf16* Ab     = take(BS * NHx * VDx);

  auto cvt = [&](const float* src, bf16* dst, size_t n) {
    int n4 = (int)(n / 4);
    cvt_kernel<<<(n4 + 255) / 256, 256, 0, stream>>>(src, dst, n4);
  };
  cvt(hs, hid_bf, BS * HIDx);
  cvt(Wq, wq_bf, (size_t)HIDx * NHx * HDx);
  cvt(Wk, wk_bf, (size_t)HIDx * NKVx * HDx);
  cvt(Wv, wv_bf, (size_t)HIDx * NKVx * VDx);
  cvt(Wo, wo_bf, (size_t)NHx * VDx * HIDx);

  gemm_bf16_tile<bf16><<<dim3((NHx * HDx) / 128, BS / 128), 256, 0, stream>>>(
      hid_bf, wq_bf, Qb, (int)BS, NHx * HDx, HIDx);
  gemm_bf16_tile<bf16><<<dim3((NKVx * HDx) / 128, BS / 128), 256, 0, stream>>>(
      hid_bf, wk_bf, Kbuf, (int)BS, NKVx * HDx, HIDx);
  gemm_bf16_tile<bf16><<<dim3((NKVx * VDx) / 128, BS / 128), 256, 0, stream>>>(
      hid_bf, wv_bf, Vb, (int)BS, NKVx * VDx, HIDx);

  rope_kernel<<<(unsigned)(BS * NHx * 32 / 256), 256, 0, stream>>>(Qb, cosb, sinb, NHx);
  rope_kernel<<<(unsigned)(BS * NKVx * 32 / 256), 256, 0, stream>>>(Kbuf, cosb, sinb, NKVx);

  attn_kernel<<<dim3(Sx / 128, NHx, Bx), 256, 0, stream>>>(Qb, Kbuf, Vb, sinks, Ab);

  gemm_bf16_tile<float><<<dim3(HIDx / 128, BS / 128), 256, 0, stream>>>(
      Ab, wo_bf, out, (int)BS, HIDx, NHx * VDx);
}
